// SelfAttention_31250182046487
// MI455X (gfx1250) — matrix operand order repaired and checked by reading, not compiled, not run on hardware
//
#include <hip/hip_runtime.h>

typedef __bf16 bf16_t;
typedef __attribute__((ext_vector_type(16))) __bf16 v16bf;
typedef __attribute__((ext_vector_type(2)))  __bf16 bf16x2;
typedef __attribute__((ext_vector_type(8)))  float  v8f;

#define SEQ 2048
#define EMB 1024
#define NH  16
#define DH  64
#define NQKV 3072   // 3 * NH * DH

// ---------------- WMMA helpers (CDNA5 gfx1250, wave32) ----------------

static __device__ __forceinline__ v8f wmma_bf16(v16bf a, v16bf b, v8f c) {
  // D = A(16x32 bf16) * B(32x16 bf16) + C(16x16 f32)
  return __builtin_amdgcn_wmma_f32_16x16x32_bf16(
      false, a, false, b, (short)0, c, false, false);
}

// A-matrix 16x32 bf16 fragment. base is [row][k] row-major with leading dim lda.
// ISA layout: lane%16 = M row; lanes<16: VGPR0..3 K=0..7 pairs, VGPR4..7 K=16..23;
// lanes>=16: +8 on the pair base.
static __device__ __forceinline__ v16bf load_a_frag(const bf16_t* base, int lda,
                                                    int row, int kb, int half) {
  v16bf a;
#pragma unroll
  for (int j = 0; j < 8; ++j) {
    int k0 = half * 8 + (j < 4 ? 2 * j : 16 + 2 * (j - 4));
    bf16x2 p = *(const bf16x2*)(base + (size_t)row * lda + kb + k0);
    a[2 * j] = p[0];
    a[2 * j + 1] = p[1];
  }
  return a;
}

// B-matrix 32x16 bf16 fragment. baseT is laid out [n][k] (contraction contiguous).
// ISA layout: N = lane%16; lanes<16: VGPR0..3 K=0..7 pairs, VGPR4..7 K=16..23; lanes>=16: +8.
static __device__ __forceinline__ v16bf load_b_frag(const bf16_t* baseT, int ldb,
                                                    int col, int kb, int half) {
  v16bf b;
#pragma unroll
  for (int v = 0; v < 8; ++v) {
    int kk = half * 8 + (v < 4 ? 2 * v : 16 + 2 * (v - 4));
    bf16x2 p = *(const bf16x2*)(baseT + (size_t)col * ldb + kb + kk);
    b[2 * v] = p[0];
    b[2 * v + 1] = p[1];
  }
  return b;
}

// ---------------- pack kernels ----------------

__global__ void pack_x(const float* __restrict__ x, bf16_t* __restrict__ xb, int n) {
  int i = blockIdx.x * blockDim.x + threadIdx.x;
  if (i < n) xb[i] = (bf16_t)x[i];
}

// WqkvT[n][e], n = proj*1024 + h*64 + d; Q weights/bias pre-scaled by 1/sqrt(Dh).
__global__ void pack_wqkv(const float* __restrict__ Wq, const float* __restrict__ Wk,
                          const float* __restrict__ Wv, const float* __restrict__ bq,
                          const float* __restrict__ bk, const float* __restrict__ bv,
                          bf16_t* __restrict__ WT, float* __restrict__ bqkv) {
  int idx = blockIdx.x * blockDim.x + threadIdx.x;
  if (idx >= NQKV * EMB) return;
  int n = idx >> 10, e = idx & 1023;
  int proj = n >> 10, h = (n >> 6) & 15, d = n & 63;
  const float* W = (proj == 0) ? Wq : (proj == 1) ? Wk : Wv;
  float sc = (proj == 0) ? 0.125f : 1.0f;  // 1/sqrt(64)
  WT[idx] = (bf16_t)(W[(((size_t)h << 10) + e) * 64 + d] * sc);
  if (e == 0) {
    const float* bb = (proj == 0) ? bq : (proj == 1) ? bk : bv;
    bqkv[n] = bb[h * 64 + d] * sc;
  }
}

__global__ void pack_wo(const float* __restrict__ Wo, bf16_t* __restrict__ WoT) {
  int idx = blockIdx.x * blockDim.x + threadIdx.x;
  if (idx >= EMB * EMB) return;
  int e = idx >> 10, f = idx & 1023;
  WoT[idx] = (bf16_t)Wo[f * EMB + e];  // WoT[e][f] = Wo[f][e]
}

// ---------------- shared 32x64-per-wave GEMM core ----------------
// acc[mi][t]: mi = M subtile (rows m0+mi*16..+15), t = N subtile (cols n0+t*16..+15)
// All 6 fragments of a K-step are loaded into distinct registers first so the
// whole load clause can stay in flight across the 8-WMMA burst.

static __device__ __forceinline__ void gemm_core_32x64(
    const bf16_t* __restrict__ A, const bf16_t* __restrict__ BT, int K,
    int m0, int n0, int ln, int half, v8f (&acc)[2][4]) {
#pragma unroll
  for (int mi = 0; mi < 2; ++mi)
#pragma unroll
    for (int t = 0; t < 4; ++t)
#pragma unroll
      for (int i = 0; i < 8; ++i) acc[mi][t][i] = 0.f;

  for (int kb = 0; kb < K; kb += 32) {
    v16bf a0 = load_a_frag(A, K, m0 + ln, kb, half);
    v16bf a1 = load_a_frag(A, K, m0 + 16 + ln, kb, half);
    v16bf b[4];
#pragma unroll
    for (int t = 0; t < 4; ++t)
      b[t] = load_b_frag(BT, K, n0 + t * 16 + ln, kb, half);
#pragma unroll
    for (int t = 0; t < 4; ++t) {
      acc[0][t] = wmma_bf16(a0, b[t], acc[0][t]);
      acc[1][t] = wmma_bf16(a1, b[t], acc[1][t]);
    }
  }
}

// ---------------- pass 1: fused QKV projection GEMM ----------------
// 2048x1024 (bf16) x 1024x3072 (bf16) -> Q/K row-major [h][s][64], V transposed [h][64][s]

__global__ __launch_bounds__(128) void qkv_gemm(
    const bf16_t* __restrict__ xb, const bf16_t* __restrict__ WT,
    const float* __restrict__ bqkv, bf16_t* __restrict__ Qm,
    bf16_t* __restrict__ Km, bf16_t* __restrict__ Vt) {
  const int lane = threadIdx.x & 31, wave = threadIdx.x >> 5;
  const int half = lane >> 4, ln = lane & 15;
  const int m0 = blockIdx.x * 128 + wave * 32;
  const int n0 = blockIdx.y * 64;
  v8f acc[2][4];
  gemm_core_32x64(xb, WT, EMB, m0, n0, ln, half, acc);

#pragma unroll
  for (int t = 0; t < 4; ++t) {
    const int n = n0 + t * 16 + ln;
    const int proj = n >> 10, h = (n >> 6) & 15, d = n & 63;
    const float bias = bqkv[n];
#pragma unroll
    for (int mi = 0; mi < 2; ++mi)
#pragma unroll
      for (int r = 0; r < 8; ++r) {
        const int s = m0 + mi * 16 + r + half * 8;
        bf16_t val = (bf16_t)(acc[mi][t][r] + bias);
        if (proj == 0)      Qm[(((size_t)h << 11) + s) * DH + d] = val;
        else if (proj == 1) Km[(((size_t)h << 11) + s) * DH + d] = val;
        else                Vt[(((size_t)(h << 6) + d) << 11) + s] = val;
      }
  }
}

// ---------------- pass 2: causal flash attention ----------------
// One wave owns 32 query rows (two 16-row M tiles sharing K/V fragments);
// iterates 32-key chunks with online softmax.

__global__ __launch_bounds__(128) void flash_attn(
    const bf16_t* __restrict__ Qm, const bf16_t* __restrict__ Km,
    const bf16_t* __restrict__ Vt, bf16_t* __restrict__ Ob) {
  __shared__ bf16_t lds[4][2][16 * 32];
  const int lane = threadIdx.x & 31, wave = threadIdx.x >> 5;
  const int half = lane >> 4, ln = lane & 15;
  const int h = blockIdx.y;
  const int q0 = (blockIdx.x * 4 + wave) * 32;
  const bf16_t* Qh = Qm + (size_t)h * SEQ * DH;
  const bf16_t* Kh = Km + (size_t)h * SEQ * DH;
  const bf16_t* Vh = Vt + (size_t)h * DH * SEQ;

  v16bf aq[2][2];  // [mi][kb]
#pragma unroll
  for (int mi = 0; mi < 2; ++mi)
#pragma unroll
    for (int kbi = 0; kbi < 2; ++kbi)
      aq[mi][kbi] = load_a_frag(Qh, DH, q0 + mi * 16 + ln, kbi * 32, half);

  const float NEG = -1e30f;
  float mrow[2][8], lrow[2][8];
  v8f o[2][4];
#pragma unroll
  for (int mi = 0; mi < 2; ++mi) {
#pragma unroll
    for (int r = 0; r < 8; ++r) { mrow[mi][r] = NEG; lrow[mi][r] = 0.f; }
#pragma unroll
    for (int t = 0; t < 4; ++t)
#pragma unroll
      for (int i = 0; i < 8; ++i) o[mi][t][i] = 0.f;
  }

  for (int c0 = 0; c0 <= q0 + 31; c0 += 32) {  // causal: skip fully-masked chunks
    // ---- S = Q * K^T : all 4 K fragments loaded up front, reused for both M tiles ----
    v16bf bk[2][2];  // [sub][kb]
#pragma unroll
    for (int sub = 0; sub < 2; ++sub) {
      bk[sub][0] = load_b_frag(Kh, DH, c0 + sub * 16 + ln, 0, half);
      bk[sub][1] = load_b_frag(Kh, DH, c0 + sub * 16 + ln, 32, half);
    }
    v8f st[2][2];  // [mi][sub]
#pragma unroll
    for (int sub = 0; sub < 2; ++sub)
#pragma unroll
      for (int mi = 0; mi < 2; ++mi) {
        v8f c;
#pragma unroll
        for (int i = 0; i < 8; ++i) c[i] = 0.f;
        c = wmma_bf16(aq[mi][0], bk[sub][0], c);
        c = wmma_bf16(aq[mi][1], bk[sub][1], c);
        st[mi][sub] = c;
      }
    // ---- causal mask (only the diagonal chunk; wave-uniform branch) ----
    if (c0 + 31 > q0) {
#pragma unroll
      for (int mi = 0; mi < 2; ++mi)
#pragma unroll
        for (int sub = 0; sub < 2; ++sub)
#pragma unroll
          for (int r = 0; r < 8; ++r) {
            int row = q0 + mi * 16 + r + half * 8;
            int col = c0 + sub * 16 + ln;
            st[mi][sub][r] = (col > row) ? NEG : st[mi][sub][r];
          }
    }
    // ---- online softmax per M tile ----
#pragma unroll
    for (int mi = 0; mi < 2; ++mi) {
      float mc[8];
#pragma unroll
      for (int r = 0; r < 8; ++r) mc[r] = fmaxf(st[mi][0][r], st[mi][1][r]);
#pragma unroll
      for (int mk = 1; mk <= 8; mk <<= 1)
#pragma unroll
        for (int r = 0; r < 8; ++r) mc[r] = fmaxf(mc[r], __shfl_xor(mc[r], mk, 32));

      float alpha[8], rs[8];
#pragma unroll
      for (int r = 0; r < 8; ++r) {
        float mn = fmaxf(mrow[mi][r], mc[r]);
        alpha[r] = __expf(mrow[mi][r] - mn);
        mrow[mi][r] = mn;
        rs[r] = 0.f;
      }
#pragma unroll
      for (int sub = 0; sub < 2; ++sub)
#pragma unroll
        for (int r = 0; r < 8; ++r) {
          float p = __expf(st[mi][sub][r] - mrow[mi][r]);
          st[mi][sub][r] = p;
          rs[r] += p;
        }
#pragma unroll
      for (int mk = 1; mk <= 8; mk <<= 1)
#pragma unroll
        for (int r = 0; r < 8; ++r) rs[r] += __shfl_xor(rs[r], mk, 32);
#pragma unroll
      for (int r = 0; r < 8; ++r) lrow[mi][r] = lrow[mi][r] * alpha[r] + rs[r];
#pragma unroll
      for (int t = 0; t < 4; ++t)
#pragma unroll
        for (int r = 0; r < 8; ++r) o[mi][t][r] *= alpha[r];

      // ---- transpose P (C layout -> A layout) through per-wave LDS ----
      bf16_t* Lw = lds[wave][mi];
#pragma unroll
      for (int sub = 0; sub < 2; ++sub)
#pragma unroll
        for (int r = 0; r < 8; ++r)
          Lw[(r + half * 8) * 32 + sub * 16 + ln] = (bf16_t)st[mi][sub][r];
    }
    asm volatile("s_wait_dscnt 0x0" ::: "memory");  // DS write->read, same wave

    v16bf ap[2];
#pragma unroll
    for (int mi = 0; mi < 2; ++mi)
#pragma unroll
      for (int j = 0; j < 8; ++j) {
        int k0 = half * 8 + (j < 4 ? 2 * j : 16 + 2 * (j - 4));
        bf16x2 p = *(const bf16x2*)(lds[wave][mi] + ln * 32 + k0);
        ap[mi][2 * j] = p[0];
        ap[mi][2 * j + 1] = p[1];
      }
    // ---- O += P * V : all 4 V fragments loaded up front, reused for both M tiles ----
    v16bf bv[4];
#pragma unroll
    for (int t = 0; t < 4; ++t)
      bv[t] = load_b_frag(Vh, SEQ, t * 16 + ln, c0, half);
#pragma unroll
    for (int t = 0; t < 4; ++t) {
      o[0][t] = wmma_bf16(ap[0], bv[t], o[0][t]);
      o[1][t] = wmma_bf16(ap[1], bv[t], o[1][t]);
    }
  }
  // ---- finalize: O /= l, concat heads into Ob[s][h*64+d] (bf16) ----
#pragma unroll
  for (int mi = 0; mi < 2; ++mi) {
#pragma unroll
    for (int r = 0; r < 8; ++r) lrow[mi][r] = 1.0f / lrow[mi][r];
#pragma unroll
    for (int t = 0; t < 4; ++t)
#pragma unroll
      for (int r = 0; r < 8; ++r) {
        int s = q0 + mi * 16 + r + half * 8;
        Ob[(size_t)s * EMB + (h << 6) + t * 16 + ln] =
            (bf16_t)(o[mi][t][r] * lrow[mi][r]);
      }
  }
}

// ---------------- pass 3: output projection GEMM + bias (fp32 out) ----------------

__global__ __launch_bounds__(128) void out_gemm(
    const bf16_t* __restrict__ Ob, const bf16_t* __restrict__ WoT,
    const float* __restrict__ bo, float* __restrict__ out) {
  const int lane = threadIdx.x & 31, wave = threadIdx.x >> 5;
  const int half = lane >> 4, ln = lane & 15;
  const int m0 = blockIdx.x * 128 + wave * 32;
  const int n0 = blockIdx.y * 64;
  v8f acc[2][4];
  gemm_core_32x64(Ob, WoT, EMB, m0, n0, ln, half, acc);

#pragma unroll
  for (int t = 0; t < 4; ++t) {
    const int n = n0 + t * 16 + ln;
    const float bias = bo[n];
#pragma unroll
    for (int mi = 0; mi < 2; ++mi)
#pragma unroll
      for (int r = 0; r < 8; ++r) {
        const int s = m0 + mi * 16 + r + half * 8;
        out[(size_t)s * EMB + n] = acc[mi][t][r] + bias;
      }
  }
}

// ---------------- host side ----------------

extern "C" void kernel_launch(void* const* d_in, const int* in_sizes, int n_in,
                              void* d_out, int out_size, void* d_ws, size_t ws_size,
                              hipStream_t stream) {
  const float* x  = (const float*)d_in[0];
  const float* Wq = (const float*)d_in[1];
  const float* bq = (const float*)d_in[2];
  const float* Wk = (const float*)d_in[3];
  const float* bk = (const float*)d_in[4];
  const float* Wv = (const float*)d_in[5];
  const float* bv = (const float*)d_in[6];
  const float* Wo = (const float*)d_in[7];
  const float* bo = (const float*)d_in[8];
  float* out = (float*)d_out;

  char* ws = (char*)d_ws;
  bf16_t* xb    = (bf16_t*)(ws);                                  // 4 MB
  bf16_t* WqkvT = (bf16_t*)(ws + ((size_t)4 << 20));              // 6 MB
  float*  bqkv  = (float*) (ws + ((size_t)10 << 20));             // 12 KB
  bf16_t* Qb    = (bf16_t*)(ws + ((size_t)10 << 20) + (64 << 10));// 4 MB
  bf16_t* Kb    = Qb + (size_t)NH * SEQ * DH;                     // 4 MB
  bf16_t* VT    = Kb + (size_t)NH * SEQ * DH;                     // 4 MB (transposed [h][d][s])
  bf16_t* Ob    = VT + (size_t)NH * SEQ * DH;                     // 4 MB
  bf16_t* WoT   = Ob + (size_t)SEQ * EMB;                         // 2 MB

  pack_x   <<<(SEQ * EMB) / 256, 256, 0, stream>>>(x, xb, SEQ * EMB);
  pack_wqkv<<<(NQKV * EMB) / 256, 256, 0, stream>>>(Wq, Wk, Wv, bq, bk, bv, WqkvT, bqkv);
  pack_wo  <<<(EMB * EMB) / 256, 256, 0, stream>>>(Wo, WoT);
  qkv_gemm <<<dim3(SEQ / 128, NQKV / 64), 128, 0, stream>>>(xb, WqkvT, bqkv, Qb, Kb, VT);
  flash_attn<<<dim3(SEQ / 128, NH), 128, 0, stream>>>(Qb, Kb, VT, Ob);
  out_gemm <<<dim3(SEQ / 128, EMB / 64), 128, 0, stream>>>(Ob, WoT, bo, out);
}
